// PriorWrapper_13623636263718
// MI455X (gfx1250) — matrix operand order repaired and checked by reading, not compiled, not run on hardware
//
#include <hip/hip_runtime.h>
#include <hip/hip_bf16.h>

// ---------------------------------------------------------------------------
// Autoregressive transformer prior sampler for MI455X (gfx1250).
// Prefill (128 tokens) uses v_wmma_f32_16x16x32_bf16 matrix cores with
// double-buffered LDS-staged B operands (batched global loads -> ds_store_b16,
// ds_load_2addr_b32 fragment reads, conflict-free 17-dword pitch) and 16x64
// register blocking per wave: 4 B fragments pinned live simultaneously ->
// one s_wait_dscnt, then 4 back-to-back WMMAs (no hazard NOPs between them).
// Decode (47 steps) is KV-cached fp32 GEMV chains (latency-bound, L2-resident).
// ---------------------------------------------------------------------------

#define D        512
#define LAT      64
#define FF       2048
#define NL       4
#define STEPS    48
#define PPRIM    128          // T // 4
#define HEADS    8
#define DH       64
#define TMAX     176          // 128 prefill + 47 decode + pad

#define EPI_BIAS  1
#define EPI_RESID 2
#define EPI_GELU  4
#define EPI_PE    8

typedef __attribute__((ext_vector_type(16))) __bf16 v16bf;
typedef __attribute__((ext_vector_type(8)))  float  v8f;
typedef __attribute__((ext_vector_type(8)))  int    v8i;

// ------------------------- device helpers ----------------------------------

__device__ __forceinline__ float pe_val(int pos, int d) {
    // pe[pos, d] = sin(pos / 10000^(2*idx/D)) for d<D/2, cos(...) for d>=D/2
    int dd = (d < (D / 2)) ? d : d - (D / 2);
    float ang = (float)pos * expf(-(2.0f * (float)dd / (float)D) * 9.210340371976184f); // ln(1e4)
    return (d < (D / 2)) ? sinf(ang) : cosf(ang);
}

__device__ __forceinline__ float gelu_tanh(float x) {
    float x3 = x * x * x;
    return 0.5f * x * (1.0f + tanhf(0.7978845608028654f * (x + 0.044715f * x3)));
}

// A-fragment 16x32 bf16 (ISA 7.12.2): lane m=L&15, half=L>>4.
// VGPR j (elems 2j,2j+1): j<4 -> K = 8*half + 2j + p ; j>=4 -> K = 16 + 8*half + 2(j-4) + p
__device__ __forceinline__ int a_kidx(int e, int half) {
    int j = e >> 1, p = e & 1;
    return (j < 4) ? (half * 8 + 2 * j + p) : (16 + half * 8 + 2 * (j - 4) + p);
}

__device__ __forceinline__ v8f wmma_bf16(v16bf a, v16bf b, v8f c) {
    return __builtin_amdgcn_wmma_f32_16x16x32_bf16(
        /*neg_a=*/false, a, /*neg_b=*/false, b,
        /*c_mod=*/(short)0, c, /*reuse_a=*/false, /*reuse_b=*/false);
}

// ------------------------- WMMA GEMM (prefill) ------------------------------
// Requires M == 128 (8 waves x 16 rows). Block computes [128] x [64] tile strip.
// B chunk (32 x 64) double-buffered in LDS as bf16, transposed [n][k] with a
// 17-dword row pitch (17 coprime 64 banks -> conflict-free stage & fragment
// reads). A elements are each touched exactly once per block -> direct global
// (pairs merge into wide loads).
#define BPITCH 34   // bf16 elements per LDS row (= 17 dwords)

__global__ __launch_bounds__(256) void wmma_gemm_k(
        const float* __restrict__ A, int lda,
        const float* __restrict__ B, int ldb,
        const float* __restrict__ bias,
        const float* __restrict__ R, int ldr,
        float* __restrict__ C, int ldc,
        int N, int K, int flags, int pos_base) {
    __shared__ __bf16 lb[2][64 * BPITCH];
    const int tid  = (int)threadIdx.x;
    const int wave = tid >> 5, lane = tid & 31;
    const int half = lane >> 4, l16 = lane & 15;
    const int nb   = (int)blockIdx.x * 64;
    const float* Arow = A + (long)(wave * 16 + l16) * lda;

    // Cooperative B stage (32x64 chunk). Per thread: fixed column n = tid&63,
    // k = tid>>6 + 4i. Load all 8 values into distinct registers (pinned) so
    // the 8 global_load_b32 stay in flight together, then convert + ds_store.
    const int sn = tid & 63, sk0 = tid >> 6;
    auto stage = [&](int kb, __bf16* dst) {
        const float* src = B + (long)(kb + sk0) * ldb + nb + sn;
        const long step = 4 * (long)ldb;
        float t0 = src[0 * step], t1 = src[1 * step], t2 = src[2 * step],
              t3 = src[3 * step], t4 = src[4 * step], t5 = src[5 * step],
              t6 = src[6 * step], t7 = src[7 * step];
        asm volatile("" : "+v"(t0), "+v"(t1), "+v"(t2), "+v"(t3),
                          "+v"(t4), "+v"(t5), "+v"(t6), "+v"(t7));
        __bf16* d = dst + sn * BPITCH + sk0;
        d[0]  = (__bf16)t0; d[4]  = (__bf16)t1; d[8]  = (__bf16)t2;
        d[12] = (__bf16)t3; d[16] = (__bf16)t4; d[20] = (__bf16)t5;
        d[24] = (__bf16)t6; d[28] = (__bf16)t7;
    };

    v8f acc0 = {}, acc1 = {}, acc2 = {}, acc3 = {};
    stage(0, lb[0]);
    __syncthreads();

    int cur = 0;
    for (int kb = 0; kb < K; kb += 32) {
        // Software pipeline: stage next chunk into the alternate buffer while
        // this chunk feeds the matrix pipe.
        if (kb + 32 < K) stage(kb + 32, lb[cur ^ 1]);

        // A fragment: paired k indices -> wide global loads.
        v16bf af;
#pragma unroll
        for (int e = 0; e < 16; ++e)
            af[e] = (__bf16)Arow[kb + a_kidx(e, half)];

        // All 4 B fragments loaded to distinct registers.
        // B layout: lane n=L&15, same K order as A: dword j<4 holds K pair
        // 8*half+2j(+1); j>=4 holds K pair 16+8*half+2(j-4)(+1).
        const int* lbi = (const int*)lb[cur];
        const int base = l16 * 17 + half * 4;
        v8i w0, w1, w2, w3;
#pragma unroll
        for (int j = 0; j < 8; ++j) {
            const int jj = (j < 4) ? j : j + 4;
            w0[j] = lbi[base + jj];
            w1[j] = lbi[base + 16 * 17 + jj];
            w2[j] = lbi[base + 32 * 17 + jj];
            w3[j] = lbi[base + 48 * 17 + jj];
        }
        // Pin all 4 fragments live simultaneously so regalloc cannot coalesce
        // them into one tuple: one s_wait_dscnt, then 4 back-to-back WMMAs.
        asm volatile("" : "+v"(w0), "+v"(w1), "+v"(w2), "+v"(w3));
        v16bf b0 = __builtin_bit_cast(v16bf, w0);
        v16bf b1 = __builtin_bit_cast(v16bf, w1);
        v16bf b2 = __builtin_bit_cast(v16bf, w2);
        v16bf b3 = __builtin_bit_cast(v16bf, w3);

        acc0 = wmma_bf16(af, b0, acc0);
        acc1 = wmma_bf16(af, b1, acc1);
        acc2 = wmma_bf16(af, b2, acc2);
        acc3 = wmma_bf16(af, b3, acc3);

        __syncthreads();        // produce->consume and consume->overwrite fence
        cur ^= 1;
    }

    // Epilogue. C/D layout: lane n=L&15, row m = 8*(L>>4) + vgpr index.
#pragma unroll
    for (int tt = 0; tt < 4; ++tt) {
        v8f acc = (tt == 0 ? acc0 : tt == 1 ? acc1 : tt == 2 ? acc2 : acc3);
        int n = nb + tt * 16 + l16;
#pragma unroll
        for (int r = 0; r < 8; ++r) {
            int m = wave * 16 + half * 8 + r;
            float v = acc[r];
            if (flags & EPI_BIAS)  v += bias[n];
            if (flags & EPI_PE)    v += pe_val(pos_base + m, n);
            if (flags & EPI_GELU)  v = gelu_tanh(v);
            if (flags & EPI_RESID) v += R[(long)m * ldr + n];
            C[(long)m * ldc + n] = v;
        }
    }
    (void)N;
}

// ------------------------- LayerNorm (block per row) ------------------------
__global__ void layernorm_k(const float* __restrict__ in,
                            const float* __restrict__ g,
                            const float* __restrict__ b,
                            float* __restrict__ out) {
    int row = (int)blockIdx.x;
    int tid = (int)threadIdx.x;           // blockDim = 256, D = 512
    const float* x = in + (long)row * D;
    float* y = out + (long)row * D;
    float v0 = x[tid], v1 = x[tid + 256];
    __shared__ float s1[256], s2[256];
    s1[tid] = v0 + v1;
    s2[tid] = v0 * v0 + v1 * v1;
    __syncthreads();
    for (int off = 128; off > 0; off >>= 1) {
        if (tid < off) { s1[tid] += s1[tid + off]; s2[tid] += s2[tid + off]; }
        __syncthreads();
    }
    float mean = s1[0] * (1.0f / D);
    float var  = s2[0] * (1.0f / D) - mean * mean;
    float rstd = rsqrtf(var + 1e-5f);
    y[tid]       = (v0 - mean) * rstd * g[tid]       + b[tid];
    y[tid + 256] = (v1 - mean) * rstd * g[tid + 256] + b[tid + 256];
}

// ------------------------- attention (prefill) ------------------------------
// One block per (query row m, head h); causal -> keys j <= m.
__global__ void attn_prefill_k(const float* __restrict__ q,
                               const float* __restrict__ kc,
                               const float* __restrict__ vc,
                               float* __restrict__ av) {
    int m = (int)blockIdx.x / HEADS;
    int h = (int)blockIdx.x % HEADS;
    int tid = (int)threadIdx.x;           // blockDim = 128
    __shared__ float p[192];
    __shared__ float red[128];
    const float* qv = q + (long)m * D + h * DH;

    float lmax = -1e30f;
    for (int j = tid; j <= m; j += 128) {
        const float* kv = kc + (long)j * D + h * DH;
        float s = 0.f;
#pragma unroll 8
        for (int d = 0; d < DH; ++d) s += qv[d] * kv[d];
        s *= 0.125f;                      // 1/sqrt(64)
        p[j] = s;
        lmax = fmaxf(lmax, s);
    }
    red[tid] = lmax; __syncthreads();
    for (int off = 64; off > 0; off >>= 1) {
        if (tid < off) red[tid] = fmaxf(red[tid], red[tid + off]);
        __syncthreads();
    }
    float mx = red[0]; __syncthreads();
    float lsum = 0.f;
    for (int j = tid; j <= m; j += 128) { float e = expf(p[j] - mx); p[j] = e; lsum += e; }
    red[tid] = lsum; __syncthreads();
    for (int off = 64; off > 0; off >>= 1) {
        if (tid < off) red[tid] += red[tid + off];
        __syncthreads();
    }
    float inv = 1.0f / red[0];
    __syncthreads();
    if (tid < DH) {
        float acc = 0.f;
        for (int j = 0; j <= m; ++j) acc += p[j] * vc[(long)j * D + h * DH + tid];
        av[(long)m * D + h * DH + tid] = acc * inv;
    }
}

// ------------------------- attention (decode, 1 query) ----------------------
__global__ void attn_decode_k(const float* __restrict__ qvec,
                              const float* __restrict__ kc,
                              const float* __restrict__ vc,
                              float* __restrict__ av, int pos) {
    int h = (int)blockIdx.x;
    int tid = (int)threadIdx.x;           // blockDim = 128
    __shared__ float p[192];
    __shared__ float red[128];
    const float* qv = qvec + h * DH;

    float lmax = -1e30f;
    for (int j = tid; j <= pos; j += 128) {
        const float* kv = kc + (long)j * D + h * DH;
        float s = 0.f;
#pragma unroll 8
        for (int d = 0; d < DH; ++d) s += qv[d] * kv[d];
        s *= 0.125f;
        p[j] = s;
        lmax = fmaxf(lmax, s);
    }
    red[tid] = lmax; __syncthreads();
    for (int off = 64; off > 0; off >>= 1) {
        if (tid < off) red[tid] = fmaxf(red[tid], red[tid + off]);
        __syncthreads();
    }
    float mx = red[0]; __syncthreads();
    float lsum = 0.f;
    for (int j = tid; j <= pos; j += 128) { float e = expf(p[j] - mx); p[j] = e; lsum += e; }
    red[tid] = lsum; __syncthreads();
    for (int off = 64; off > 0; off >>= 1) {
        if (tid < off) red[tid] += red[tid + off];
        __syncthreads();
    }
    float inv = 1.0f / red[0];
    __syncthreads();
    if (tid < DH) {
        float acc = 0.f;
        for (int j = 0; j <= pos; ++j) acc += p[j] * vc[(long)j * D + h * DH + tid];
        av[h * DH + tid] = acc * inv;
    }
}

// ------------------------- GEMV (decode steps) -------------------------------
// y[n] = epi( sum_k x[k]*W[k*ldw+n] + bias[n] + pe + add2[n] ) (+ resid[n])
// Coalesced across n (row-major weights); fp32 throughout.
__global__ void gemv_k(const float* __restrict__ x,
                       const float* __restrict__ W, int ldw,
                       const float* __restrict__ bias,
                       const float* __restrict__ resid,
                       const float* __restrict__ add2,
                       float* __restrict__ y,
                       float* __restrict__ y2, int y2_stride,
                       int N, int K, int flags, int pos) {
    int n = (int)(blockIdx.x * blockDim.x + threadIdx.x);
    if (n >= N) return;
    float acc = bias ? bias[n] : 0.f;
    for (int k = 0; k < K; ++k) acc += x[k] * W[(long)k * ldw + n];
    if (flags & EPI_PE)   acc += pe_val(pos, n);
    if (add2)             acc += add2[n];
    if (flags & EPI_GELU) acc = gelu_tanh(acc);
    if (resid)            acc += resid[n];
    y[n] = acc;
    if (y2) y2[(long)n * y2_stride] = acc;
}

// ---------------------------------------------------------------------------
extern "C" void kernel_launch(void* const* d_in, const int* in_sizes, int n_in,
                              void* d_out, int out_size, void* d_ws, size_t ws_size,
                              hipStream_t stream) {
    (void)in_sizes; (void)n_in; (void)out_size; (void)ws_size;
    // setup_inputs() order:
    const float* prior  = (const float*)d_in[1];   // [1, 512, 64]
    const float* in_w   = (const float*)d_in[2];   // [64, 512]
    const float* in_b   = (const float*)d_in[3];   // [512]
    const float* qkv_w  = (const float*)d_in[4];   // [4, 512, 1536]
    const float* qkv_b  = (const float*)d_in[5];   // [4, 1536]
    const float* o_w    = (const float*)d_in[6];   // [4, 512, 512]
    const float* o_b    = (const float*)d_in[7];   // [4, 512]
    const float* ln1_g  = (const float*)d_in[8];
    const float* ln1_b  = (const float*)d_in[9];
    const float* ln2_g  = (const float*)d_in[10];
    const float* ln2_b  = (const float*)d_in[11];
    const float* ff1_w  = (const float*)d_in[12];  // [4, 512, 2048]
    const float* ff1_b  = (const float*)d_in[13];  // [4, 2048]
    const float* ff2_w  = (const float*)d_in[14];  // [4, 2048, 512]
    const float* ff2_b  = (const float*)d_in[15];  // [4, 512]
    const float* lnf_g  = (const float*)d_in[16];
    const float* lnf_b  = (const float*)d_in[17];
    const float* out_w  = (const float*)d_in[18];  // [512, 64]
    const float* out_b  = (const float*)d_in[19];  // [64]
    const float* noise  = (const float*)d_in[20];  // [48, 64]
    float* dout = (float*)d_out;                   // [1, 64, 48] -> dout[l*48 + t]

    // Workspace carve-up (all fp32; ~4.7 MB)
    float* ws  = (float*)d_ws;
    float* hb  = ws;                        // [128][512] hidden states (prefill)
    float* ab  = hb  + 128 * D;             // [128][512] LN output
    float* qb  = ab  + 128 * D;             // [128][512] q (prefill)
    float* avb = qb  + 128 * D;             // [128][512] attention output
    float* ub  = avb + 128 * D;             // [128][2048] FF hidden
    float* kc  = ub  + 128 * FF;            // [4][176][512] K cache
    float* vc  = kc  + NL * TMAX * D;       // [4][176][512] V cache
    float* xv  = vc  + NL * TMAX * D;       // [512] decode hidden
    float* avv = xv  + D;                   // [512] decode LN out
    float* qd  = avv + D;                   // [512] decode q
    float* avd = qd  + D;                   // [512] decode attn out
    float* fd  = avd + D;                   // [2048] decode FF hidden
    float* lat = fd  + FF;                  // [64] current latent

    // M is always 128 (8 waves x 16 rows per block); grid tiles N in strips of 64.
    auto gemm = [&](const float* A, int lda, const float* B, int ldb,
                    const float* bias, const float* R, int ldr,
                    float* C, int ldc, int N, int K, int flags, int pos) {
        wmma_gemm_k<<<N / 64, 256, 0, stream>>>(A, lda, B, ldb, bias, R, ldr,
                                                C, ldc, N, K, flags, pos);
    };

    // ---------------- Prefill: rows 0..127 through all layers ---------------
    // embed: h = buf @ in_w + in_b + pe   (M=128, K=64, N=512)
    gemm(prior, LAT, in_w, D, in_b, nullptr, 0, hb, D, D, LAT, EPI_BIAS | EPI_PE, 0);

    for (int l = 0; l < NL; ++l) {
        const float* qw = qkv_w + (long)l * D * 3 * D;
        layernorm_k<<<128, 256, 0, stream>>>(hb, ln1_g + l * D, ln1_b + l * D, ab);
        // q / k / v as three N=512 GEMMs; K,V land directly in the cache.
        gemm(ab, D, qw,         3 * D, qkv_b + l * 3 * D,         nullptr, 0,
             qb, D, D, D, EPI_BIAS, 0);
        gemm(ab, D, qw + D,     3 * D, qkv_b + l * 3 * D + D,     nullptr, 0,
             kc + (long)l * TMAX * D, D, D, D, EPI_BIAS, 0);
        gemm(ab, D, qw + 2 * D, 3 * D, qkv_b + l * 3 * D + 2 * D, nullptr, 0,
             vc + (long)l * TMAX * D, D, D, D, EPI_BIAS, 0);
        attn_prefill_k<<<128 * HEADS, 128, 0, stream>>>(
            qb, kc + (long)l * TMAX * D, vc + (long)l * TMAX * D, avb);
        // h += av @ o_w + o_b
        gemm(avb, D, o_w + (long)l * D * D, D, o_b + l * D, hb, D,
             hb, D, D, D, EPI_BIAS | EPI_RESID, 0);
        layernorm_k<<<128, 256, 0, stream>>>(hb, ln2_g + l * D, ln2_b + l * D, ab);
        gemm(ab, D, ff1_w + (long)l * D * FF, FF, ff1_b + l * FF, nullptr, 0,
             ub, FF, FF, D, EPI_BIAS | EPI_GELU, 0);
        gemm(ub, FF, ff2_w + (long)l * FF * D, D, ff2_b + l * D, hb, D,
             hb, D, D, FF, EPI_BIAS | EPI_RESID, 0);
    }

    // Step t=0: head on row P-1 = 127
    layernorm_k<<<1, 256, 0, stream>>>(hb + (long)127 * D, lnf_g, lnf_b, avv);
    gemv_k<<<1, 64, 0, stream>>>(avv, out_w, LAT, out_b, nullptr, noise + 0,
                                 lat, dout + 0, STEPS, LAT, D, 0, 0);

    // ---------------- Decode: steps t = 1..47, KV-cached ---------------------
    for (int t = 1; t < STEPS; ++t) {
        int pos = PPRIM + t - 1;            // position of token latent_{t-1}
        // embed: x = latent @ in_w + in_b + pe[pos]
        gemv_k<<<2, 256, 0, stream>>>(lat, in_w, D, in_b, nullptr, nullptr,
                                      xv, nullptr, 0, D, LAT, EPI_PE, pos);
        for (int l = 0; l < NL; ++l) {
            const float* qw = qkv_w + (long)l * D * 3 * D;
            layernorm_k<<<1, 256, 0, stream>>>(xv, ln1_g + l * D, ln1_b + l * D, avv);
            gemv_k<<<2, 256, 0, stream>>>(avv, qw,         3 * D, qkv_b + l * 3 * D,
                                          nullptr, nullptr, qd, nullptr, 0, D, D, 0, 0);
            gemv_k<<<2, 256, 0, stream>>>(avv, qw + D,     3 * D, qkv_b + l * 3 * D + D,
                                          nullptr, nullptr,
                                          kc + (long)l * TMAX * D + (long)pos * D,
                                          nullptr, 0, D, D, 0, 0);
            gemv_k<<<2, 256, 0, stream>>>(avv, qw + 2 * D, 3 * D, qkv_b + l * 3 * D + 2 * D,
                                          nullptr, nullptr,
                                          vc + (long)l * TMAX * D + (long)pos * D,
                                          nullptr, 0, D, D, 0, 0);
            attn_decode_k<<<HEADS, 128, 0, stream>>>(
                qd, kc + (long)l * TMAX * D, vc + (long)l * TMAX * D, avd, pos);
            gemv_k<<<2, 256, 0, stream>>>(avd, o_w + (long)l * D * D, D, o_b + l * D,
                                          xv, nullptr, xv, nullptr, 0, D, D, 0, 0);
            layernorm_k<<<1, 256, 0, stream>>>(xv, ln2_g + l * D, ln2_b + l * D, avv);
            gemv_k<<<8, 256, 0, stream>>>(avv, ff1_w + (long)l * D * FF, FF, ff1_b + l * FF,
                                          nullptr, nullptr, fd, nullptr, 0, FF, D, EPI_GELU, 0);
            gemv_k<<<2, 256, 0, stream>>>(fd, ff2_w + (long)l * FF * D, D, ff2_b + l * D,
                                          xv, nullptr, xv, nullptr, 0, D, FF, 0, 0);
        }
        layernorm_k<<<1, 256, 0, stream>>>(xv, lnf_g, lnf_b, avv);
        gemv_k<<<1, 64, 0, stream>>>(avv, out_w, LAT, out_b, nullptr, noise + (long)t * LAT,
                                     lat, dout + t, STEPS, LAT, D, 0, 0);
    }
}
